// SelfAttentionLayer_2740189135747
// MI455X (gfx1250) — matrix operand order repaired and checked by reading, not compiled, not run on hardware
//
#include <hip/hip_runtime.h>
#include <math.h>

#define S_DIM 4096
#define D_DIM 512

typedef __attribute__((ext_vector_type(16))) __bf16 v16bf;
typedef __attribute__((ext_vector_type(8)))  float  v8f;

__device__ __forceinline__ unsigned short f2bf(float f) {
    union { float f; unsigned u; } c; c.f = f;
    unsigned u = c.u;
    unsigned r = u + 0x7FFFu + ((u >> 16) & 1u);   // round-to-nearest-even
    return (unsigned short)(r >> 16);
}

// ---------------- elementwise f32 -> bf16 ----------------
__global__ void cvt_bf16_kernel(const float* __restrict__ in,
                                unsigned short* __restrict__ out, int n) {
    int i = blockIdx.x * blockDim.x + threadIdx.x;
    if (i < n) out[i] = f2bf(in[i]);
}

// ---------------- bf16 transpose [rows x cols] -> [cols x rows] ----------------
__global__ void transpose_bf16_kernel(const unsigned short* __restrict__ in,
                                      unsigned short* __restrict__ out,
                                      int rows, int cols) {
    __shared__ unsigned short tile[32][33];
    int c0 = blockIdx.x * 32;
    int r0 = blockIdx.y * 32;
    int tx = threadIdx.x, ty = threadIdx.y;   // block (32, 8)
    #pragma unroll
    for (int j = 0; j < 32; j += 8)
        tile[ty + j][tx] = in[(size_t)(r0 + ty + j) * cols + (c0 + tx)];
    __syncthreads();
    #pragma unroll
    for (int j = 0; j < 32; j += 8)
        out[(size_t)(c0 + ty + j) * rows + (r0 + tx)] = tile[tx][ty + j];
}

// ---------------- tiled bf16 WMMA GEMM: C[M,N] = A[M,K] * Bt[N,K]^T ----------------
// EPI 0: store bf16 to Cb.  EPI 1: val*scale, mask==0 -> -1e10, store f32 (N==S).
// EPI 2: store f32.
template <int EPI>
__global__ __launch_bounds__(256)
void gemm_bf16_kernel(const unsigned short* __restrict__ A,
                      const unsigned short* __restrict__ Bt,
                      int K, int N,
                      float* __restrict__ Cf,
                      unsigned short* __restrict__ Cb,
                      const int* __restrict__ mask,
                      float scale) {
    const int LDSTR = 40;  // 32 bf16 + 8 pad (80B rows, 16B aligned)
    __shared__ __attribute__((aligned(16))) unsigned short ldsA[128 * LDSTR];
    __shared__ __attribute__((aligned(16))) unsigned short ldsB[128 * LDSTR];

    const int tid  = threadIdx.x;
    const int lane = tid & 31;
    const int wid  = tid >> 5;   // 0..7
    const int wm   = wid & 3;    // wave row 0..3  -> 32 rows of M each
    const int wn   = wid >> 2;   // wave col 0..1  -> 64 cols of N each
    const int m0   = blockIdx.y * 128;
    const int n0   = blockIdx.x * 128;

    const int ldrow  = tid >> 1;  // 0..127: tile row this thread stages
    const int ldhalf = tid & 1;   // which 16-element K half

    const int r  = lane & 15;
    const int hi = lane >> 4;

    v8f acc[2][4];
    #pragma unroll
    for (int i = 0; i < 2; ++i)
        #pragma unroll
        for (int j = 0; j < 4; ++j)
            acc[i][j] = (v8f){0.f, 0.f, 0.f, 0.f, 0.f, 0.f, 0.f, 0.f};

    for (int kt = 0; kt < K; kt += 32) {
        // stage 128x32 bf16 tiles of A and Bt into LDS (one 32B chunk per thread per tile)
        {
            const unsigned short* ga = A + (size_t)(m0 + ldrow) * K + kt + ldhalf * 16;
            uint4 a0 = *(const uint4*)ga;
            uint4 a1 = *(const uint4*)(ga + 8);
            unsigned short* sa = &ldsA[ldrow * LDSTR + ldhalf * 16];
            *(uint4*)sa       = a0;
            *(uint4*)(sa + 8) = a1;

            const unsigned short* gb = Bt + (size_t)(n0 + ldrow) * K + kt + ldhalf * 16;
            uint4 b0 = *(const uint4*)gb;
            uint4 b1 = *(const uint4*)(gb + 8);
            unsigned short* sb = &ldsB[ldrow * LDSTR + ldhalf * 16];
            *(uint4*)sb       = b0;
            *(uint4*)(sb + 8) = b1;
        }
        __syncthreads();

        union FragU { v16bf v; uint4 u[2]; };
        FragU afrag[2], bfrag[4];
        // A fragment: lane<16 holds K{0..7,16..23}, lane>=16 holds K{8..15,24..31}
        #pragma unroll
        for (int i = 0; i < 2; ++i) {
            const unsigned short* pa = &ldsA[(wm * 32 + i * 16 + r) * LDSTR + hi * 8];
            afrag[i].u[0] = *(const uint4*)pa;
            afrag[i].u[1] = *(const uint4*)(pa + 16);
        }
        // B fragment: lane<16 = column r with K{0..7,16..23}; lane>=16 K{8..15,24..31}
        #pragma unroll
        for (int j = 0; j < 4; ++j) {
            const unsigned short* pb = &ldsB[(wn * 64 + j * 16 + r) * LDSTR + hi * 8];
            bfrag[j].u[0] = *(const uint4*)pb;
            bfrag[j].u[1] = *(const uint4*)(pb + 16);
        }
        #pragma unroll
        for (int i = 0; i < 2; ++i)
            #pragma unroll
            for (int j = 0; j < 4; ++j)
                acc[i][j] = __builtin_amdgcn_wmma_f32_16x16x32_bf16(
                    false, afrag[i].v, false, bfrag[j].v,
                    (short)0, acc[i][j], false, false);
        __syncthreads();
    }

    // epilogue: lane = N column, VGPR v + hi*8 = M row within 16x16 tile
    #pragma unroll
    for (int i = 0; i < 2; ++i) {
        #pragma unroll
        for (int j = 0; j < 4; ++j) {
            const int cn = n0 + wn * 64 + j * 16 + r;
            const int rb = m0 + wm * 32 + i * 16 + hi * 8;
            #pragma unroll
            for (int v = 0; v < 8; ++v) {
                float val = acc[i][j][v];
                const size_t idx = (size_t)(rb + v) * N + cn;
                if (EPI == 0) {
                    Cb[idx] = f2bf(val);
                } else if (EPI == 1) {
                    val *= scale;
                    if (mask[idx] == 0) val = -1e10f;   // N == S for this epilogue
                    Cf[idx] = val;
                } else {
                    Cf[idx] = val;
                }
            }
        }
    }
}

// ---------------- row softmax over [S,S]; also emit bf16 copy ----------------
__global__ __launch_bounds__(256)
void softmax_kernel(float* __restrict__ attn, unsigned short* __restrict__ attnb) {
    const int row = blockIdx.x;
    const int tid = threadIdx.x;
    float* p = attn + (size_t)row * S_DIM;

    float v[16];
    float mx = -3.4e38f;
    #pragma unroll
    for (int i = 0; i < 16; ++i) {
        v[i] = p[tid + i * 256];
        mx = fmaxf(mx, v[i]);
    }
    __shared__ float red[256];
    red[tid] = mx;
    __syncthreads();
    for (int off = 128; off > 0; off >>= 1) {
        if (tid < off) red[tid] = fmaxf(red[tid], red[tid + off]);
        __syncthreads();
    }
    const float m = red[0];
    __syncthreads();

    float s = 0.f;
    #pragma unroll
    for (int i = 0; i < 16; ++i) {
        v[i] = __expf(v[i] - m);
        s += v[i];
    }
    red[tid] = s;
    __syncthreads();
    for (int off = 128; off > 0; off >>= 1) {
        if (tid < off) red[tid] += red[tid + off];
        __syncthreads();
    }
    const float inv = 1.0f / red[0];

    unsigned short* pb = attnb + (size_t)row * S_DIM;
    #pragma unroll
    for (int i = 0; i < 16; ++i) {
        float o = v[i] * inv;
        p[tid + i * 256]  = o;
        pb[tid + i * 256] = f2bf(o);
    }
}

extern "C" void kernel_launch(void* const* d_in, const int* in_sizes, int n_in,
                              void* d_out, int out_size, void* d_ws, size_t ws_size,
                              hipStream_t stream) {
    const int S = S_DIM, D = D_DIM;
    const float* query = (const float*)d_in[0];
    const float* key   = (const float*)d_in[1];
    const float* value = (const float*)d_in[2];
    const int*   mask  = (const int*)d_in[3];
    const float* w_q   = (const float*)d_in[4];
    const float* w_k   = (const float*)d_in[5];
    const float* w_v   = (const float*)d_in[6];

    float* x_out = (float*)d_out;               // [S,D]
    float* attn  = x_out + (size_t)S * D;       // [S,S]

    unsigned short* p = (unsigned short*)d_ws;
    const size_t DD = (size_t)D * D;
    const size_t SD = (size_t)S * D;
    unsigned short* wqb = p;
    unsigned short* wkb = wqb + DD;
    unsigned short* wvb = wkb + DD;
    unsigned short* Qb  = wvb + DD;             // [S,D] bf16
    unsigned short* Kb  = Qb + SD;              // [S,D] bf16
    unsigned short* Vt  = Kb + SD;              // [D,S] bf16
    unsigned short* scratch = Vt + SD;
    unsigned short* qb = scratch;               // dead after projections
    unsigned short* kb = qb + SD;
    unsigned short* vb = kb + SD;
    unsigned short* Vb = vb + SD;               // dead after transpose
    unsigned short* attnb = scratch;            // [S,S] bf16, reuses dead region

    const float scale = 1.0f / sqrtf((float)D);

    // 1) convert inputs to bf16
    {
        int n1 = S * D, n2 = D * D;
        cvt_bf16_kernel<<<(n1 + 255) / 256, 256, 0, stream>>>(query, qb, n1);
        cvt_bf16_kernel<<<(n1 + 255) / 256, 256, 0, stream>>>(key,   kb, n1);
        cvt_bf16_kernel<<<(n1 + 255) / 256, 256, 0, stream>>>(value, vb, n1);
        cvt_bf16_kernel<<<(n2 + 255) / 256, 256, 0, stream>>>(w_q, wqb, n2);
        cvt_bf16_kernel<<<(n2 + 255) / 256, 256, 0, stream>>>(w_k, wkb, n2);
        cvt_bf16_kernel<<<(n2 + 255) / 256, 256, 0, stream>>>(w_v, wvb, n2);
    }
    // 2) projections: Q = q @ wq^T, K = k @ wk^T, V = v @ wv^T  (store bf16)
    {
        dim3 grid(D / 128, S / 128);
        gemm_bf16_kernel<0><<<grid, 256, 0, stream>>>(qb, wqb, D, D, nullptr, Qb, nullptr, 1.0f);
        gemm_bf16_kernel<0><<<grid, 256, 0, stream>>>(kb, wkb, D, D, nullptr, Kb, nullptr, 1.0f);
        gemm_bf16_kernel<0><<<grid, 256, 0, stream>>>(vb, wvb, D, D, nullptr, Vb, nullptr, 1.0f);
    }
    // 3) Vt = V^T (so GEMM3's B operand is K-contiguous)
    {
        dim3 grid(D / 32, S / 32);
        transpose_bf16_kernel<<<grid, dim3(32, 8), 0, stream>>>(Vb, Vt, S, D);
    }
    // 4) energy = (Q @ K^T)/sqrt(D), masked -> f32 attention buffer
    {
        dim3 grid(S / 128, S / 128);
        gemm_bf16_kernel<1><<<grid, 256, 0, stream>>>(Qb, Kb, D, S, attn, nullptr, mask, scale);
    }
    // 5) softmax rows (f32 in d_out, bf16 copy for next GEMM)
    softmax_kernel<<<S, 256, 0, stream>>>(attn, attnb);
    // 6) x = attention @ V
    {
        dim3 grid(D / 128, S / 128);
        gemm_bf16_kernel<2><<<grid, 256, 0, stream>>>(attnb, Vt, S, D, x_out, nullptr, nullptr, 1.0f);
    }
}
